// CantorGlobalAttention_73297911874092
// MI455X (gfx1250) — matrix operand order repaired and checked by reading, not compiled, not run on hardware
//
#include <hip/hip_runtime.h>
#include <hip/hip_bf16.h>
#include <stdint.h>

typedef _Float16 v8h  __attribute__((ext_vector_type(8)));
typedef _Float16 v16h __attribute__((ext_vector_type(16)));
typedef float    v8f  __attribute__((ext_vector_type(8)));
typedef int      v4i  __attribute__((vector_size(16)));   // matches builtin param type

#define N_DIR 5
#define NE    16
#define NB    8
#define NP    256
#define ND    128
#define NW    3
#define NK    768            // W * P
#define CK    32             // K-chunk = WMMA K for f16
#define RP    128            // rows of P handled per block
#define NCHUNK (NK / CK)     // 24
#define LSTR  40             // padded row stride (halves) for A / Vt: conflict-free b128 loads
#define EPSV  1e-6f

#if defined(__has_builtin)
#if __has_builtin(__builtin_amdgcn_global_load_async_to_lds_b128) && \
    __has_builtin(__builtin_amdgcn_s_wait_asynccnt)
#define USE_ASYNC 1
#endif
#endif
#ifndef USE_ASYNC
#define USE_ASYNC 0
#endif

struct SharedMem {
  float s[NK];                       // gated, temperature-scaled neighbor K affinities
  float q[RP];
  float m[RP];                       // analytic row max
  float zs[RP];                      // w_dir / Z_p
  float zpart[256];
  float redmin[256];
  float redmax[256];
  alignas(16) _Float16 A[RP * LSTR]; // A tile (f16), row-major, padded
  alignas(16) _Float16 Vt[ND * LSTR];// V^T tile (f16): [d][k], padded
#if USE_ASYNC
  alignas(16) float raw[2][CK * ND]; // double-buffered raw f32 V chunks (async landing zone)
#endif
};

__global__ __launch_bounds__(256)
void cantor_attn_kernel(const float* __restrict__ Qa,
                        const float* __restrict__ Ka,
                        const float* __restrict__ Vp,
                        const float* __restrict__ betas,
                        const float* __restrict__ temp,
                        const float* __restrict__ fw,
                        const int*   __restrict__ routes,
                        float* __restrict__ out)
{
  __shared__ SharedMem sm;
  const int t    = threadIdx.x;
  const int lane = t & 31;
  const int wave = t >> 5;
  const int gid  = blockIdx.x;        // [0, 256)
  const int e    = gid >> 4;
  const int b    = (gid >> 1) & 7;
  const int half = gid & 1;           // which 128-row half of P

  const float invt = 1.0f / (fabsf(temp[0]) + EPSV);

  // fusion softmax (cheap, per-thread)
  float fwv[N_DIR];
  float fmax = -3.4e38f;
  #pragma unroll
  for (int d = 0; d < N_DIR; ++d) { fwv[d] = fw[d]; fmax = fmaxf(fmax, fwv[d]); }
  float fsum = 0.f;
  #pragma unroll
  for (int d = 0; d < N_DIR; ++d) { fwv[d] = __expf(fwv[d] - fmax); fsum += fwv[d]; }
  const float finv = 1.0f / fsum;

  const v8f vzero = {0.f, 0.f, 0.f, 0.f, 0.f, 0.f, 0.f, 0.f};
  v8f acc[8];
  #pragma unroll
  for (int i = 0; i < 8; ++i) acc[i] = vzero;

  for (int dir = 0; dir < N_DIR; ++dir) {
    const float wfd = fwv[dir] * finv;

    // ---- phase 1: gated neighbor affinities s[k], plus min/max reduction ----
    float lmin = 3.4e38f, lmax = -3.4e38f;
    for (int idx = t; idx < NK; idx += 256) {
      const int w  = idx >> 8;
      const int pp = idx & 255;
      const int re = routes[e * NW + w];
      const float bet = (re == e) ? 1.0f
                                  : (1.0f / (1.0f + __expf(-betas[e * NE + re])));
      const float val = Ka[(((size_t)dir * NE + re) * NB + b) * NP + pp] * bet * invt;
      sm.s[idx] = val;
      lmin = fminf(lmin, val);
      lmax = fmaxf(lmax, val);
    }
    sm.redmin[t] = lmin;
    sm.redmax[t] = lmax;
    __syncthreads();
    for (int off = 128; off > 0; off >>= 1) {
      if (t < off) {
        sm.redmin[t] = fminf(sm.redmin[t], sm.redmin[t + off]);
        sm.redmax[t] = fmaxf(sm.redmax[t], sm.redmax[t + off]);
      }
      __syncthreads();
    }
    const float smin = sm.redmin[0];
    const float smax = sm.redmax[0];

    // ---- phase 2: per-row q, analytic max m, partition function Z ----
    {
      const int p  = t & 127;
      const int kh = t >> 7;
      const float q = Qa[(((size_t)dir * NE + e) * NB + b) * NP + half * RP + p];
      const float m = (q >= 0.f) ? q * smax : q * smin;   // exact row max -> exp arg <= 0
      if (kh == 0) { sm.q[p] = q; sm.m[p] = m; }
      float z = 0.f;
      const int k0 = kh * (NK / 2);
      for (int k = k0; k < k0 + NK / 2; ++k)
        z += __expf(q * sm.s[k] - m);
      sm.zpart[t] = z;
    }
    __syncthreads();
    if (t < RP) {
      const float Z = sm.zpart[t] + sm.zpart[t + 128];    // Z >= 1 always
      sm.zs[t] = wfd / Z;
    }
    __syncthreads();

#if USE_ASYNC
    { // prologue: async-stage chunk 0 of this dir's V into raw[0]
      const int re0 = routes[e * NW + 0];
      const float* gV0 = Vp + (((size_t)dir * NE + re0) * NB + b) * NP * ND;
      #pragma unroll
      for (int i = 0; i < 4; ++i)
        __builtin_amdgcn_global_load_async_to_lds_b128(
            (v4i*)(gV0 + t * 4 + i * 1024),
            (v4i*)(&sm.raw[0][t * 4 + i * 1024]), 0, 0);
    }
#endif

    // ---- phase 3: K-chunk loop -> build A (f16) + V^T (f16), run WMMA ----
    for (int c = 0; c < NCHUNK; ++c) {
      const int kbase = c * CK;
      const int wsel  = kbase >> 8;
      const int re    = routes[e * NW + wsel];
      const float* gV = Vp + ((((size_t)dir * NE + re) * NB + b) * NP + (kbase & 255)) * ND;
      (void)gV;

#if USE_ASYNC
      __builtin_amdgcn_s_wait_asynccnt(0);
#endif
      __syncthreads();   // prev WMMA reads done; async data visible block-wide

      // generate A tile: 128 rows x 32 k, scaled by w_dir / Z  (values in [0,1] -> f16 safe)
      {
        const int p   = t >> 1;
        const int kh2 = t & 1;
        const float q  = sm.q[p];
        const float m  = sm.m[p];
        const float zs = sm.zs[p];
        #pragma unroll
        for (int j = 0; j < 16; ++j) {
          const int kk = kh2 * 16 + j;
          const float a = __expf(q * sm.s[kbase + kk] - m) * zs;
          sm.A[p * LSTR + kk] = (_Float16)a;
        }
      }
      // stage V chunk as transposed f16: Vt[d][k]
      {
        const int d     = t & 127;
        const int khalf = t >> 7;
        v8h h0, h1;
        #pragma unroll
        for (int j = 0; j < 8; ++j) {
          const int k = khalf * 16 + j;
#if USE_ASYNC
          h0[j] = (_Float16)sm.raw[c & 1][k * ND + d];
#else
          h0[j] = (_Float16)gV[(size_t)k * ND + d];
#endif
        }
        #pragma unroll
        for (int j = 0; j < 8; ++j) {
          const int k = khalf * 16 + 8 + j;
#if USE_ASYNC
          h1[j] = (_Float16)sm.raw[c & 1][k * ND + d];
#else
          h1[j] = (_Float16)gV[(size_t)k * ND + d];
#endif
        }
        *(v8h*)&sm.Vt[d * LSTR + khalf * 16]     = h0;
        *(v8h*)&sm.Vt[d * LSTR + khalf * 16 + 8] = h1;
      }
      __syncthreads();   // tiles ready

#if USE_ASYNC
      if (c + 1 < NCHUNK) {  // pipeline: async-stage next chunk while WMMA runs
        const int kb2 = (c + 1) * CK;
        const int re2 = routes[e * NW + (kb2 >> 8)];
        const float* gV2 = Vp + ((((size_t)dir * NE + re2) * NB + b) * NP + (kb2 & 255)) * ND;
        #pragma unroll
        for (int i = 0; i < 4; ++i)
          __builtin_amdgcn_global_load_async_to_lds_b128(
              (v4i*)(gV2 + t * 4 + i * 1024),
              (v4i*)(&sm.raw[(c + 1) & 1][t * 4 + i * 1024]), 0, 0);
      }
#else
      if (c + 1 < NCHUNK) {  // L2 prefetch of next chunk (global_prefetch_b8)
        const int kb2 = (c + 1) * CK;
        const int re2 = routes[e * NW + (kb2 >> 8)];
        const float* gV2 = Vp + ((((size_t)dir * NE + re2) * NB + b) * NP + (kb2 & 255)) * ND;
        __builtin_prefetch(gV2 + t * 16, 0, 0);
      }
#endif

      // WMMA: each wave -> 16 rows x 128 cols (8 tiles), K-step 32
      {
        const int row = wave * 16 + (lane & 15);
        const int hi  = (lane >> 4) * 8;                  // ISA 16-bit A 16x32 layout
        const v8h a0 = *(const v8h*)&sm.A[row * LSTR + hi];
        const v8h a1 = *(const v8h*)&sm.A[row * LSTR + 16 + hi];
        const v16h af = __builtin_shufflevector(a0, a1,
            0,1,2,3,4,5,6,7,8,9,10,11,12,13,14,15);
        #pragma unroll
        for (int dt = 0; dt < 8; ++dt) {
          const int dcol = dt * 16 + (lane & 15);
          const int kst  = (lane >> 4) * 8;               // B 32x16: lanes 16-31 hold K 8..15, 24..31
          const v8h b0 = *(const v8h*)&sm.Vt[dcol * LSTR + kst];
          const v8h b1 = *(const v8h*)&sm.Vt[dcol * LSTR + kst + 16];
          const v16h bf = __builtin_shufflevector(b0, b1,
              0,1,2,3,4,5,6,7,8,9,10,11,12,13,14,15);
          acc[dt] = __builtin_amdgcn_wmma_f32_16x16x32_f16(
              false, af, false, bf, (short)0, acc[dt], false, false);
        }
      }
    } // chunks
  } // dirs

  // ---- epilogue: accumulators already normalized and fusion-weighted ----
  // C/D layout: VGPR v, lane L -> M = v + 8*(L>=16), N = L&15
  const int rowbase = e * NP + half * RP + wave * 16;
  const int mhi  = (lane >> 4) * 8;
  const int ncol = lane & 15;
  #pragma unroll
  for (int dt = 0; dt < 8; ++dt) {
    #pragma unroll
    for (int v = 0; v < 8; ++v) {
      const int row = rowbase + v + mhi;
      out[((size_t)b * (NE * NP) + row) * ND + dt * 16 + ncol] = acc[dt][v];
    }
  }
}

extern "C" void kernel_launch(void* const* d_in, const int* in_sizes, int n_in,
                              void* d_out, int out_size, void* d_ws, size_t ws_size,
                              hipStream_t stream) {
  (void)in_sizes; (void)n_in; (void)d_ws; (void)ws_size; (void)out_size;
  const float* Qa     = (const float*)d_in[0];
  const float* Ka     = (const float*)d_in[1];
  const float* Vp     = (const float*)d_in[2];
  const float* betas  = (const float*)d_in[3];
  const float* temp   = (const float*)d_in[4];
  const float* fwv    = (const float*)d_in[5];
  const int*   routes = (const int*)d_in[6];

  dim3 grid(NE * NB * 2);   // (e, b, half-of-P) = 256 blocks
  dim3 block(256);          // 8 wave32 waves
  hipLaunchKernelGGL(cantor_attn_kernel, grid, block, 0, stream,
                     Qa, Ka, Vp, betas, temp, fwv, routes, (float*)d_out);
}
